// MaskedMultiHeadAttention_45595372814917
// MI455X (gfx1250) — matrix operand order repaired and checked by reading, not compiled, not run on hardware
//
#include <hip/hip_runtime.h>
#include <hip/hip_bf16.h>
#include <math.h>
#include <stdint.h>

typedef __attribute__((ext_vector_type(16))) __bf16 v16bf;
typedef __attribute__((ext_vector_type(8)))  float  v8f;

#define D_MODEL 768
#define E_DIM   2304      // 3*D
#define SEQ     2048
#define NBATCH  2
#define NHEADS  12
#define HDIM    64

static __device__ inline v8f wmma_bf16(v16bf a, v16bf b, v8f c) {
    // (neg_a, A, neg_b, B, c_mod, C, reuse_a, reuse_b)
    return __builtin_amdgcn_wmma_f32_16x16x32_bf16(false, a, false, b, (short)0, c, false, false);
}

// =====================================================================
// Kernel 1: z[m,e] = sum_k x[m,k] * W[e,k] + b[e], stored as bf16
//   x: [4096, 768] f32, W: [2304, 768] f32, z: [4096, 2304] bf16
//   Block tile 128(M) x 128(E) x 32(K); 8 waves; wave tile 32x64.
// =====================================================================
#define BM 128
#define BN 128
#define BK 32
#define K1_THREADS 256
#define K1_STRIDE 40          // 32 + 8 pad (bf16 elements)

__global__ __launch_bounds__(K1_THREADS)
void qkv_gemm_kernel(const float* __restrict__ x,
                     const float* __restrict__ w,
                     const float* __restrict__ bias,
                     __bf16* __restrict__ z)
{
    __shared__ __bf16 Xs[BM][K1_STRIDE];
    __shared__ __bf16 Ws[BN][K1_STRIDE];

    const int tid    = threadIdx.x;
    const int lane   = tid & 31;
    const int lane16 = lane & 15;
    const int half   = lane >> 4;
    const int wave   = tid >> 5;          // 0..7
    const int waveM  = wave & 3;          // 4 groups of 32 rows
    const int waveN  = wave >> 2;         // 2 groups of 64 cols

    const int rowBase = blockIdx.y * BM;  // M tile base
    const int colBase = blockIdx.x * BN;  // E tile base

    v8f acc[2][4] = {};                   // [m-subtile][n-subtile], f32 accum

    for (int k0 = 0; k0 < D_MODEL; k0 += BK) {
        // -------- stage f32 -> bf16 tiles into LDS (float4 loads) --------
        #pragma unroll
        for (int t = 0; t < 4; ++t) {
            int vec = tid + t * K1_THREADS;    // 0..1023
            int r   = vec >> 3;                // 0..127
            int c4  = (vec & 7) << 2;          // 0..28
            float4 xv = *(const float4*)&x[(size_t)(rowBase + r) * D_MODEL + k0 + c4];
            Xs[r][c4 + 0] = (__bf16)xv.x;  Xs[r][c4 + 1] = (__bf16)xv.y;
            Xs[r][c4 + 2] = (__bf16)xv.z;  Xs[r][c4 + 3] = (__bf16)xv.w;
            float4 wv = *(const float4*)&w[(size_t)(colBase + r) * D_MODEL + k0 + c4];
            Ws[r][c4 + 0] = (__bf16)wv.x;  Ws[r][c4 + 1] = (__bf16)wv.y;
            Ws[r][c4 + 2] = (__bf16)wv.z;  Ws[r][c4 + 3] = (__bf16)wv.w;
        }
        __syncthreads();

        // speculative prefetch of next K tile (global_prefetch_b8)
        if (k0 + BK < D_MODEL) {
            if (tid < 128)
                __builtin_prefetch(&x[(size_t)(rowBase + tid) * D_MODEL + k0 + BK], 0, 3);
            else
                __builtin_prefetch(&w[(size_t)(colBase + tid - 128) * D_MODEL + k0 + BK], 0, 3);
        }

        // -------- assemble fragments (ISA 7.12.2 wave32 layouts) --------
        v16bf afrag[2];
        #pragma unroll
        for (int s = 0; s < 2; ++s) {
            int row = waveM * 32 + s * 16 + lane16;
            int b1 = 8 * half, b2 = 16 + 8 * half;
            #pragma unroll
            for (int i = 0; i < 8; ++i) {
                afrag[s][i]     = Xs[row][b1 + i];
                afrag[s][8 + i] = Xs[row][b2 + i];
            }
        }
        v16bf bfrag[4];
        #pragma unroll
        for (int nt = 0; nt < 4; ++nt) {
            int rowW = waveN * 64 + nt * 16 + lane16;  // B[k][n] = W[n][k]
            #pragma unroll
            for (int i = 0; i < 16; ++i)
                bfrag[nt][i] = Ws[rowW][(i & 8) * 2 + 8 * half + (i & 7)];
        }

        #pragma unroll
        for (int s = 0; s < 2; ++s)
            #pragma unroll
            for (int nt = 0; nt < 4; ++nt)
                acc[s][nt] = wmma_bf16(afrag[s], bfrag[nt], acc[s][nt]);

        __syncthreads();
    }

    // -------- bias add, convert to bf16, store z --------
    #pragma unroll
    for (int nt = 0; nt < 4; ++nt) {
        int e = colBase + waveN * 64 + nt * 16 + lane16;
        float bv = bias[e];
        #pragma unroll
        for (int s = 0; s < 2; ++s) {
            #pragma unroll
            for (int v = 0; v < 8; ++v) {
                int m = rowBase + waveM * 32 + s * 16 + v + 8 * half;
                z[(size_t)m * E_DIM + e] = (__bf16)(acc[s][nt][v] + bv);
            }
        }
    }
}

// =====================================================================
// Kernel 2: causal flash attention over z (bf16), out f32
//   grid: (C/64, H, N); block: 128 threads = 4 waves, 16 query rows/wave
//   K tile staged with async global->LDS copies (ASYNCcnt path).
// =====================================================================
#define BQ 64
#define BKV 64
#define K2_THREADS 128
#define K2_STRIDE 72          // 64 + 8 pad (bf16 elements); 144B row, 16B aligned

__global__ __launch_bounds__(K2_THREADS)
void attention_kernel(const __bf16* __restrict__ z,
                      float* __restrict__ out)
{
    __shared__ __bf16 Qs[BQ][K2_STRIDE];
    __shared__ __bf16 Ks[BKV][K2_STRIDE];
    __shared__ __bf16 Vt[HDIM][K2_STRIDE];        // V transposed: Vt[d][key]
    __shared__ __bf16 Ps[4][16][K2_STRIDE];       // per-wave P relayout buffer

    const int tid    = threadIdx.x;
    const int lane   = tid & 31;
    const int lane16 = lane & 15;
    const int half   = lane >> 4;
    const int wave   = tid >> 5;                  // 0..3
    const int qb = blockIdx.x, h = blockIdx.y, n = blockIdx.z;

    const size_t rowOff = (size_t)n * SEQ;
    const __bf16* qbase = z + (rowOff + (size_t)qb * BQ) * E_DIM + D_MODEL + h * HDIM;
    const __bf16* kbase = z + rowOff * E_DIM + h * HDIM;
    const __bf16* vbase = z + rowOff * E_DIM + 2 * D_MODEL + h * HDIM;

    // ---- load Q tile (64x64 bf16) ----
    #pragma unroll
    for (int t = 0; t < 4; ++t) {
        int vec = tid + t * K2_THREADS;           // 0..511
        int r   = vec >> 3;                       // 0..63
        int c8  = (vec & 7) << 3;                 // 0..56
        *(uint4*)&Qs[r][c8] = *(const uint4*)(qbase + (size_t)r * E_DIM + c8);
    }
    __syncthreads();

    // ---- Q fragments: A 16x32 layout, 2 K-chunks of hd=64 ----
    v16bf qf[2];
    #pragma unroll
    for (int c = 0; c < 2; ++c) {
        int row = wave * 16 + lane16;
        #pragma unroll
        for (int i = 0; i < 8; ++i) {
            qf[c][i]     = Qs[row][c * 32 + 8 * half + i];
            qf[c][8 + i] = Qs[row][c * 32 + 16 + 8 * half + i];
        }
    }

    v8f o[4] = {};
    float m_i[8], l_i[8];
    #pragma unroll
    for (int v = 0; v < 8; ++v) { m_i[v] = -INFINITY; l_i[v] = 0.0f; }

    const float sc = 0.03608439182435161f;        // 1/sqrt(768) (full D, per source)

    for (int kb = 0; kb <= qb; ++kb) {
        __syncthreads();                          // previous tile fully consumed

        // ---- stage K tile: async global->LDS byte copies (ASYNCcnt) ----
        #pragma unroll
        for (int t = 0; t < 4; ++t) {
            int vec = tid + t * K2_THREADS;       // 0..511
            int r   = vec >> 3;
            int c8  = (vec & 7) << 3;
            const __bf16* gp = kbase + (size_t)(kb * BKV + r) * E_DIM + c8;
            unsigned ldsa = (unsigned)(uintptr_t)&Ks[r][c8];
            asm volatile("global_load_async_to_lds_b128 %0, %1, off"
                         :: "v"(ldsa), "v"(gp) : "memory");
        }

        // ---- stage transposed V tile through VGPRs (needs transpose) ----
        #pragma unroll
        for (int t = 0; t < 4; ++t) {
            int vec = tid + t * K2_THREADS;
            int r   = vec >> 3;
            int c8  = (vec & 7) << 3;
            uint4 vv = *(const uint4*)(vbase + (size_t)(kb * BKV + r) * E_DIM + c8);
            const __bf16* vp = (const __bf16*)&vv;
            #pragma unroll
            for (int j = 0; j < 8; ++j) Vt[c8 + j][r] = vp[j];
        }

        asm volatile("s_wait_asynccnt 0x0" ::: "memory");
        __syncthreads();

        // ---- S = Q K^T (8 WMMAs) ----
        v8f s[4] = {};
        #pragma unroll
        for (int c = 0; c < 2; ++c) {
            #pragma unroll
            for (int nt = 0; nt < 4; ++nt) {
                v16bf kf;                         // B[k=d][n=key] = Ks[key][d]
                int rowK = nt * 16 + lane16;
                #pragma unroll
                for (int i = 0; i < 16; ++i)
                    kf[i] = Ks[rowK][c * 32 + (i & 8) * 2 + 8 * half + (i & 7)];
                s[nt] = wmma_bf16(qf[c], kf, s[nt]);
            }
        }

        // ---- scale + causal mask (diagonal block only) ----
        const bool diag = (kb == qb);
        #pragma unroll
        for (int nt = 0; nt < 4; ++nt) {
            int key = nt * 16 + lane16;
            #pragma unroll
            for (int v = 0; v < 8; ++v) {
                float val = s[nt][v] * sc;
                int qrow = wave * 16 + v + 8 * half;
                if (diag && key > qrow) val = -INFINITY;
                s[nt][v] = val;
            }
        }

        // ---- online softmax per row (row lives in 16 lanes of one half) ----
        #pragma unroll
        for (int v = 0; v < 8; ++v) {
            float t = fmaxf(fmaxf(s[0][v], s[1][v]), fmaxf(s[2][v], s[3][v]));
            t = fmaxf(t, __shfl_xor(t, 1));
            t = fmaxf(t, __shfl_xor(t, 2));
            t = fmaxf(t, __shfl_xor(t, 4));
            t = fmaxf(t, __shfl_xor(t, 8));
            float mnew = fmaxf(m_i[v], t);
            float corr = __expf(m_i[v] - mnew);
            float rsum = 0.0f;
            #pragma unroll
            for (int nt = 0; nt < 4; ++nt) {
                float p = __expf(s[nt][v] - mnew);
                s[nt][v] = p;
                rsum += p;
            }
            rsum += __shfl_xor(rsum, 1);
            rsum += __shfl_xor(rsum, 2);
            rsum += __shfl_xor(rsum, 4);
            rsum += __shfl_xor(rsum, 8);
            l_i[v] = l_i[v] * corr + rsum;
            m_i[v] = mnew;
            #pragma unroll
            for (int nt = 0; nt < 4; ++nt) o[nt][v] *= corr;
        }

        // ---- relayout P (C-layout -> A-layout) through per-wave LDS ----
        #pragma unroll
        for (int nt = 0; nt < 4; ++nt)
            #pragma unroll
            for (int v = 0; v < 8; ++v)
                Ps[wave][v + 8 * half][nt * 16 + lane16] = (__bf16)s[nt][v];

        // ---- O += P V (8 WMMAs) ----
        #pragma unroll
        for (int c = 0; c < 2; ++c) {
            v16bf pf;
            #pragma unroll
            for (int i = 0; i < 8; ++i) {
                pf[i]     = Ps[wave][lane16][c * 32 + 8 * half + i];
                pf[8 + i] = Ps[wave][lane16][c * 32 + 16 + 8 * half + i];
            }
            #pragma unroll
            for (int nt = 0; nt < 4; ++nt) {
                v16bf vf;                         // B[k=key][n=d] = Vt[d][key]
                #pragma unroll
                for (int i = 0; i < 16; ++i)
                    vf[i] = Vt[nt * 16 + lane16][c * 32 + (i & 8) * 2 + 8 * half + (i & 7)];
                o[nt] = wmma_bf16(pf, vf, o[nt]);
            }
        }
    }

    // ---- normalize and store: out[n, qrow, h*64 + d] (f32) ----
    float* obase = out + (rowOff + (size_t)qb * BQ + wave * 16) * D_MODEL + h * HDIM;
    #pragma unroll
    for (int nt = 0; nt < 4; ++nt) {
        #pragma unroll
        for (int v = 0; v < 8; ++v) {
            int row = v + 8 * half;
            obase[(size_t)row * D_MODEL + nt * 16 + lane16] = o[nt][v] / l_i[v];
        }
    }
}

// =====================================================================
extern "C" void kernel_launch(void* const* d_in, const int* in_sizes, int n_in,
                              void* d_out, int out_size, void* d_ws, size_t ws_size,
                              hipStream_t stream) {
    const float* x    = (const float*)d_in[0];   // [2,2048,768]
    const float* Wkqv = (const float*)d_in[1];   // [2304,768]
    const float* bkqv = (const float*)d_in[2];   // [2304]
    float* out = (float*)d_out;                  // [2,2048,768]

    __bf16* z = (__bf16*)d_ws;                   // [4096, 2304] bf16 ≈ 18.9 MB

    dim3 g1(E_DIM / BN, (NBATCH * SEQ) / BM, 1);     // 18 x 32
    qkv_gemm_kernel<<<g1, K1_THREADS, 0, stream>>>(x, Wkqv, bkqv, z);

    dim3 g2(SEQ / BQ, NHEADS, NBATCH);               // 32 x 12 x 2
    attention_kernel<<<g2, K2_THREADS, 0, stream>>>(z, out);
}
